// rnn_two_crf_seq_12695923327225
// MI455X (gfx1250) — matrix operand order repaired and checked by reading, not compiled, not run on hardware
//
#include <hip/hip_runtime.h>
#include <hip/hip_bf16.h>

typedef unsigned short u16;
typedef __attribute__((ext_vector_type(16))) __bf16 v16bf;
typedef __attribute__((ext_vector_type(8)))  __bf16 v8bf;
typedef __attribute__((ext_vector_type(8)))  float  v8f;

#define Tn   512
#define Bn   64
#define Ln   513
#define TBn  (Tn*Bn)
#define HDh  512
#define G4H  2048
#define NEGV (-10000.0f)

// ---------- bf16 helpers (round-to-nearest-even) ----------
__device__ __forceinline__ u16 f2bf_u(float f) {
    unsigned u = __builtin_bit_cast(unsigned, f);
    unsigned r = u + 0x7FFFu + ((u >> 16) & 1u);
    return (u16)(r >> 16);
}
__device__ __forceinline__ float bf2f_u(u16 h) {
    unsigned u = ((unsigned)h) << 16;
    return __builtin_bit_cast(float, u);
}
__device__ __forceinline__ float sigm(float x) { return 1.0f / (1.0f + __expf(-x)); }

__device__ __forceinline__ v8f z8() {
    v8f v;
#pragma unroll
    for (int j = 0; j < 8; ++j) v[j] = 0.0f;
    return v;
}

// A-fragment: 16x32 bf16 tile row for this lane (contiguous K layout).
// p points at row start + kk; lane-half 'hi' selects K 0..15 or 16..31.
__device__ __forceinline__ v16bf load_afrag(const u16* p, int hi) {
    const v8bf* q = reinterpret_cast<const v8bf*>(p + hi * 16);
    v8bf lo = q[0];
    v8bf hp = q[1];        // +8 elements
    v16bf r;
#pragma unroll
    for (int j = 0; j < 8; ++j) { r[j] = lo[j]; r[j + 8] = hp[j]; }
    return r;
}

__device__ __forceinline__ v8f wmma_bf16(v16bf a, v16bf b, v8f c) {
    return __builtin_amdgcn_wmma_f32_16x16x32_bf16(false, a, false, b, (short)0, c, false, false);
}

// ---------- small prep kernels ----------
__global__ void pack_bf16_kernel(const float* __restrict__ src, u16* __restrict__ dst,
                                 int rows, int sc, int dc) {
    long id = (long)blockIdx.x * blockDim.x + threadIdx.x;
    if (id >= (long)rows * dc) return;
    int c = (int)(id % dc);
    long r = id / dc;
    dst[id] = (c < sc) ? f2bf_u(src[r * sc + c]) : (u16)0;
}

__global__ void addv_kernel(const float* a, const float* b, float* o, int n) {
    int id = blockIdx.x * blockDim.x + threadIdx.x;
    if (id < n) o[id] = a[id] + b[id];
}

__global__ void embed_kernel(const int* __restrict__ xw, const float* __restrict__ We,
                             u16* __restrict__ x0) {
    long id = (long)blockIdx.x * blockDim.x + threadIdx.x; // TBn*128 threads, 4 elems each
    if (id >= (long)TBn * 128) return;
    int q = (int)(id & 127);
    long r = id >> 7;                 // r = t*64 + b
    int b = (int)(r & 63);
    int t = (int)(r >> 6);
    int tok = xw[b * Ln + t];
    float4 v = *reinterpret_cast<const float4*>(We + (size_t)tok * 512 + q * 4);
    u16* dst = x0 + r * 512 + q * 4;
    dst[0] = f2bf_u(v.x); dst[1] = f2bf_u(v.y);
    dst[2] = f2bf_u(v.z); dst[3] = f2bf_u(v.w);
}

// ---------- P = X @ Wih^T + (bih+bhh), WMMA GEMM, M=T*B, N=2048 ----------
__global__ __launch_bounds__(256) void gemm_p_kernel(
        const u16* __restrict__ Xin, const u16* __restrict__ Wih,
        const float* __restrict__ bias, u16* __restrict__ P, int DinPad) {
    int lane = threadIdx.x & 31, wave = threadIdx.x >> 5;
    int hi = lane >> 4, r15 = lane & 15;
    int dir = blockIdx.z;
    int colbase = blockIdx.x * 128 + wave * 16;
    long rowbase = (long)blockIdx.y * 64;

    const u16* Wd = Wih + (size_t)dir * G4H * DinPad;
    const u16* brow = Wd + (size_t)(colbase + r15) * DinPad;
    const u16* arow0 = Xin + (size_t)(rowbase + r15) * DinPad;

    v8f acc[4];
#pragma unroll
    for (int m = 0; m < 4; ++m) acc[m] = z8();

    for (int kk = 0; kk < DinPad; kk += 32) {
        v16bf bf = *reinterpret_cast<const v16bf*>(brow + kk + hi * 16);
#pragma unroll
        for (int mt = 0; mt < 4; ++mt) {
            v16bf af = load_afrag(arow0 + (size_t)mt * 16 * DinPad + kk, hi);
            acc[mt] = wmma_bf16(af, bf, acc[mt]);
        }
    }
    float bcol = bias[dir * G4H + colbase + r15];
    u16* Pd = P + (size_t)dir * TBn * G4H;
#pragma unroll
    for (int mt = 0; mt < 4; ++mt)
#pragma unroll
        for (int e = 0; e < 8; ++e) {
            long rr = rowbase + mt * 16 + e + 8 * hi;
            Pd[rr * G4H + colbase + r15] = f2bf_u(acc[mt][e] + bcol);
        }
}

// ---------- recurrent LSTM step kernel ----------
// grid = (16 hidden-slices, 2 directions). Each workgroup owns j in [jbase,jbase+32),
// keeps its 128x512 bf16 Whh slice in LDS for all 512 steps, cell state in VGPRs,
// and synchronizes per-step with a device-scope atomic barrier per direction.
__global__ __launch_bounds__(256) void lstm_rec_kernel(
        const u16* __restrict__ P, const u16* __restrict__ Whh,
        u16* __restrict__ hglob, u16* __restrict__ Out, unsigned* __restrict__ bar) {
    extern __shared__ __align__(256) char smem[];
    u16* ldsw = (u16*)smem;           // [128][512] bf16 = 128 KB

    int tid = threadIdx.x, lane = tid & 31, wave = tid >> 5;
    int hi = lane >> 4, r15 = lane & 15;
    int dir = blockIdx.y;
    int jbase = blockIdx.x * 32;

    // Stage Whh slice (rows g*512 + jbase + jl, g=0..3, jl=0..31) into LDS.
    const u16* Wd = Whh + (size_t)dir * G4H * HDh;
    for (int idx = tid; idx < 8192; idx += 256) {            // 8192 uint4 = 128 KB
        int lr = idx >> 6;
        int ec = (idx & 63) * 8;
        int g = lr >> 5, jl = lr & 31;
        uint4 v = *reinterpret_cast<const uint4*>(Wd + (size_t)(g * 512 + jbase + jl) * HDh + ec);
        *reinterpret_cast<uint4*>(ldsw + (size_t)lr * HDh + ec) = v;
    }
    __syncthreads();

    int mt = wave >> 1, jt = wave & 1;         // 4 m-tiles x 2 j-tiles over 8 waves
    int jloc = jt * 16 + r15;
    int j = jbase + jloc;

    float cst[8];
#pragma unroll
    for (int e = 0; e < 8; ++e) cst[e] = 0.0f;

    const u16* Pd = P + (size_t)dir * TBn * G4H;
    unsigned* mybar = bar + dir * 16;          // separate cachelines per direction

    for (int tt = 0; tt < Tn; ++tt) {
        int t = dir ? (Tn - 1 - tt) : tt;
        int cur = tt & 1, prev = cur ^ 1;      // tt=0 reads buffer 1 (pre-zeroed)
        const u16* hprev = hglob + (size_t)(dir * 2 + prev) * Bn * HDh;

        v8f acc[4];
#pragma unroll
        for (int g = 0; g < 4; ++g) acc[g] = z8();

        const u16* arow = hprev + (size_t)(mt * 16 + r15) * HDh;
        for (int kk = 0; kk < HDh; kk += 32) {
            v16bf af = load_afrag(arow + kk, hi);
#pragma unroll
            for (int g = 0; g < 4; ++g) {
                v16bf bf = *reinterpret_cast<const v16bf*>(
                    ldsw + (size_t)(g * 32 + jloc) * HDh + kk + hi * 16);
                acc[g] = wmma_bf16(af, bf, acc[g]);
            }
        }

        u16* hcur = hglob + (size_t)(dir * 2 + cur) * Bn * HDh;
#pragma unroll
        for (int e = 0; e < 8; ++e) {
            int m = mt * 16 + e + 8 * hi;
            const u16* pr = Pd + ((size_t)t * Bn + m) * G4H;
            float iv = acc[0][e] + bf2f_u(pr[0 * 512 + j]);
            float fv = acc[1][e] + bf2f_u(pr[1 * 512 + j]);
            float gv = acc[2][e] + bf2f_u(pr[2 * 512 + j]);
            float ov = acc[3][e] + bf2f_u(pr[3 * 512 + j]);
            float cn = sigm(fv) * cst[e] + sigm(iv) * tanhf(gv);
            cst[e] = cn;
            float hv = sigm(ov) * tanhf(cn);
            u16 hb = f2bf_u(hv);
            hcur[(size_t)m * HDh + j] = hb;
            Out[((size_t)t * Bn + m) * 1024 + dir * 512 + j] = hb;
        }

        // device-wide step barrier (16 workgroups per direction)
        __threadfence();
        __syncthreads();
        if (tid == 0) {
            __hip_atomic_fetch_add(mybar, 1u, __ATOMIC_RELEASE, __HIP_MEMORY_SCOPE_AGENT);
            unsigned tgt = 16u * (unsigned)(tt + 1);
            while (__hip_atomic_load(mybar, __ATOMIC_ACQUIRE, __HIP_MEMORY_SCOPE_AGENT) < tgt)
                __builtin_amdgcn_s_sleep(1);
        }
        __syncthreads();
    }
}

// ---------- output projections (K = 12 or 24, tiny) ----------
__global__ void proj_kernel(const u16* __restrict__ In, const float* __restrict__ W,
                            const float* __restrict__ bias, const int* __restrict__ xw,
                            float* __restrict__ Outp, int K, int applyMask) {
    long id = (long)blockIdx.x * blockDim.x + threadIdx.x;
    if (id >= (long)TBn * K) return;
    int k = (int)(id % K);
    long r = id / K;
    const u16* row = In + r * 1024;
    const float* wr = W + (size_t)k * 1024;
    float s = 0.0f;
    for (int i = 0; i < 1024; ++i) s += bf2f_u(row[i]) * wr[i];
    s += bias[k];
    if (applyMask) {
        int b = (int)(r & 63);
        int t = (int)(r >> 6);
        if (!(xw[b * Ln + t + 1] > 0)) s = 0.0f;
    }
    Outp[id] = s;
}

__global__ void xner_pad_kernel(const float* __restrict__ hiob, u16* __restrict__ xner) {
    long id = (long)blockIdx.x * blockDim.x + threadIdx.x;
    if (id >= (long)TBn * 32) return;
    int k = (int)(id & 31);
    long r = id >> 5;
    xner[id] = (k < 12) ? f2bf_u(hiob[r * 12 + k]) : (u16)0;
}

// ---------- CRF forward (partition) : one wave per batch row ----------
__global__ void crf_part_kernel(const float* __restrict__ h, const float* __restrict__ trans,
                                const int* __restrict__ xw, float* __restrict__ Z, int K) {
    int b = blockIdx.x;
    int lane = threadIdx.x;
    bool act = lane < K;
    float score = (lane == 1) ? 0.0f : NEGV;   // SOS_IDX = 1
    for (int t = 0; t < Tn; ++t) {
        float mt = (xw[b * Ln + t + 1] > 0) ? 1.0f : 0.0f;
        float mx = -3.0e38f;
        for (int jj = 0; jj < K; ++jj) {
            float sj = __shfl(score, jj, 32);
            float v = sj + (act ? trans[lane * K + jj] : 0.0f);
            mx = fmaxf(mx, v);
        }
        float sum = 0.0f;
        for (int jj = 0; jj < K; ++jj) {
            float sj = __shfl(score, jj, 32);
            float v = sj + (act ? trans[lane * K + jj] : 0.0f);
            sum += __expf(v - mx);
        }
        float hv = act ? h[((size_t)t * Bn + b) * K + lane] : 0.0f;
        float s = hv + mx + __logf(sum);
        float ns = mt * s + (1.0f - mt) * score;
        score = act ? ns : NEGV;
    }
    float v = act ? (score + trans[2 * K + lane]) : -3.0e38f;  // EOS row
    float mx = v;
    for (int o = 16; o > 0; o >>= 1) mx = fmaxf(mx, __shfl_xor(mx, o, 32));
    float e = __expf(v - mx);
    for (int o = 16; o > 0; o >>= 1) e += __shfl_xor(e, o, 32);
    if (lane == 0) Z[b] = mx + __logf(e);
}

// ---------- CRF gold score : one thread per batch row ----------
__global__ void crf_gold_kernel(const float* __restrict__ h, const int* __restrict__ y,
                                const float* __restrict__ trans, const int* __restrict__ xw,
                                float* __restrict__ S, int K) {
    int b = threadIdx.x;
    if (b >= Bn) return;
    float sc = 0.0f;
    int len = 0;
    for (int t = 0; t < Tn; ++t) {
        if (xw[b * Ln + t + 1] > 0) {
            int yi = y[b * Ln + t + 1];
            int yp = y[b * Ln + t];
            sc += h[((size_t)t * Bn + b) * K + yi] + trans[yi * K + yp];
            len++;
        }
    }
    int last = y[b * Ln + len];
    sc += trans[2 * K + last];
    S[b] = sc;
}

__global__ void final_kernel(const float* Zi, const float* Si,
                             const float* Zn, const float* Sn, float* out) {
    if (threadIdx.x == 0) {
        float a = 0.0f, c = 0.0f;
        for (int b = 0; b < Bn; ++b) { a += Zi[b] - Si[b]; c += Zn[b] - Sn[b]; }
        out[0] = a / 64.0f;
        out[1] = c / 64.0f;
    }
}

// ----------------------------------------------------------------------
extern "C" void kernel_launch(void* const* d_in, const int* in_sizes, int n_in,
                              void* d_out, int out_size, void* d_ws, size_t ws_size,
                              hipStream_t stream) {
    (void)in_sizes; (void)n_in; (void)out_size; (void)ws_size;

    const int*   xw     = (const int*)d_in[1];
    const int*   yiob   = (const int*)d_in[2];
    const int*   yner   = (const int*)d_in[3];
    const float* Wemb   = (const float*)d_in[4];
    const float* WoutI  = (const float*)d_in[21];
    const float* boutI  = (const float*)d_in[22];
    const float* trI    = (const float*)d_in[23];
    const float* WoutN  = (const float*)d_in[40];
    const float* boutN  = (const float*)d_in[41];
    const float* trN    = (const float*)d_in[42];
    float* out = (float*)d_out;

    // ---- workspace bump allocator (assumes ws is large; ~460 MB used) ----
    char* w = (char*)d_ws;
    size_t off = 0;
    auto alloc = [&](size_t bytes) -> char* {
        off = (off + 255) & ~(size_t)255;
        char* p = w + off;
        off += bytes;
        return p;
    };

    u16* x0   = (u16*)alloc((size_t)TBn * 512 * 2);
    u16* h1   = (u16*)alloc((size_t)TBn * 1024 * 2);
    u16* h2   = (u16*)alloc((size_t)TBn * 1024 * 2);
    u16* Pbuf = (u16*)alloc((size_t)2 * TBn * G4H * 2);

    const int wbase[4] = {5, 13, 24, 32};
    const int srcC[4]  = {512, 1024, 12, 1024};
    const int dpad[4]  = {512, 1024, 32, 1024};
    u16*   wih_bf[4]; u16* whh_bf[4]; float* bcb[4];
    for (int s = 0; s < 4; ++s) {
        wih_bf[s] = (u16*)alloc((size_t)2 * G4H * dpad[s] * 2);
        whh_bf[s] = (u16*)alloc((size_t)2 * G4H * HDh * 2);
        bcb[s]    = (float*)alloc((size_t)2 * G4H * 4);
    }
    float* hiob   = (float*)alloc((size_t)TBn * 12 * 4);
    u16*   xner   = (u16*)alloc((size_t)TBn * 32 * 2);
    float* nerout = (float*)alloc((size_t)TBn * 24 * 4);
    u16*   hglob  = (u16*)alloc((size_t)2 * 2 * Bn * HDh * 2);
    unsigned* ctrl = (unsigned*)alloc(256);
    float* Zi = (float*)alloc(64 * 4);
    float* Si = (float*)alloc(64 * 4);
    float* Zn = (float*)alloc(64 * 4);
    float* Sn = (float*)alloc(64 * 4);

    // ---- weight conversion to bf16 (+combined biases) ----
    for (int s = 0; s < 4; ++s) {
        for (int d = 0; d < 2; ++d) {
            const float* Wih = (const float*)d_in[wbase[s] + d * 4 + 0];
            const float* Whh = (const float*)d_in[wbase[s] + d * 4 + 1];
            const float* bih = (const float*)d_in[wbase[s] + d * 4 + 2];
            const float* bhh = (const float*)d_in[wbase[s] + d * 4 + 3];
            long n1 = (long)G4H * dpad[s];
            pack_bf16_kernel<<<(unsigned)((n1 + 255) / 256), 256, 0, stream>>>(
                Wih, wih_bf[s] + (size_t)d * G4H * dpad[s], G4H, srcC[s], dpad[s]);
            long n2 = (long)G4H * HDh;
            pack_bf16_kernel<<<(unsigned)((n2 + 255) / 256), 256, 0, stream>>>(
                Whh, whh_bf[s] + (size_t)d * G4H * HDh, G4H, HDh, HDh);
            addv_kernel<<<(G4H + 255) / 256, 256, 0, stream>>>(bih, bhh, bcb[s] + d * G4H, G4H);
        }
    }

    // ---- embedding ----
    {
        long n = (long)TBn * 128;
        embed_kernel<<<(unsigned)((n + 255) / 256), 256, 0, stream>>>(xw, Wemb, x0);
    }

    auto run_stage = [&](int s, const u16* Xin, u16* Out) {
        hipMemsetAsync(ctrl, 0, 256, stream);
        hipMemsetAsync(hglob, 0, (size_t)2 * 2 * Bn * HDh * 2, stream);
        gemm_p_kernel<<<dim3(16, 512, 2), 256, 0, stream>>>(Xin, wih_bf[s], bcb[s], Pbuf, dpad[s]);
        lstm_rec_kernel<<<dim3(16, 2), 256, 131072, stream>>>(Pbuf, whh_bf[s], hglob, Out, ctrl);
    };

    // ---- IOB BiLSTM stack ----
    run_stage(0, x0, h1);
    run_stage(1, h1, h2);
    {
        long n = (long)TBn * 12;
        proj_kernel<<<(unsigned)((n + 255) / 256), 256, 0, stream>>>(h2, WoutI, boutI, xw, hiob, 12, 1);
        long m = (long)TBn * 32;
        xner_pad_kernel<<<(unsigned)((m + 255) / 256), 256, 0, stream>>>(hiob, xner);
    }

    // ---- NER BiLSTM stack (reuses h1/h2) ----
    run_stage(2, xner, h1);
    run_stage(3, h1, h2);
    {
        long n = (long)TBn * 24;
        proj_kernel<<<(unsigned)((n + 255) / 256), 256, 0, stream>>>(h2, WoutN, boutN, xw, nerout, 24, 0);
    }

    // ---- CRF partition + gold + final reduction ----
    crf_part_kernel<<<Bn, 32, 0, stream>>>(hiob, trI, xw, Zi, 12);
    crf_part_kernel<<<Bn, 32, 0, stream>>>(nerout, trN, xw, Zn, 24);
    crf_gold_kernel<<<1, 64, 0, stream>>>(hiob, yiob, trI, xw, Si, 12);
    crf_gold_kernel<<<1, 64, 0, stream>>>(nerout, yner, trN, xw, Sn, 24);
    final_kernel<<<1, 32, 0, stream>>>(Zi, Si, Zn, Sn, out);
}
